// TransolverBlock_70815420776670
// MI455X (gfx1250) — matrix operand order repaired and checked by reading, not compiled, not run on hardware
//
#include <hip/hip_runtime.h>

// ============================================================================
// Transolver block for MI455X (gfx1250, wave32, WMMA).
//
// Roofline: ~196 GFLOP, ~0.4 GB traffic -> compute-dominated unless matrix
// ops run on WMMA. All eight large GEMMs (Wfx, Wx, Wslice-logits, st-einsum,
// out_x-einsum, Wo, W1, W2) use v_wmma_f32_16x16x32_bf16 with f32 accum.
// 192MB L2 keeps weights + 32MB activation tiles resident, so fragments are
// loaded directly from global as b128 loads (no LDS staging in the GEMM).
// Everything is deterministic: no atomics, fixed-order reductions.
// ============================================================================

typedef __bf16 bf16;
typedef __bf16 v16bf __attribute__((ext_vector_type(16)));
typedef float  v8f   __attribute__((ext_vector_type(8)));
typedef unsigned u32x4 __attribute__((ext_vector_type(4)));

union Frag { v16bf v; u32x4 q[2]; };

__device__ __forceinline__ float gelu_f(float x) {
    return 0.5f * x * (1.0f + erff(x * 0.70710678118654752f));
}

// ---------------------------------------------------------------------------
// dims
// ---------------------------------------------------------------------------
#define BB 4
#define NN 8192
#define DD 512
#define HH 8
#define GG 64
#define DHH 64
#define DFF 2048
#define SZ (BB*NN)          // 32768 rows

// ---------------------------------------------------------------------------
// Generic batched GEMM:  C[m,n] = sum_k A[m,k] * Bt[n,k]  (+bias[n]) (+gelu)
// A: bf16 row-major, lda; Bt: bf16 with rows contiguous in k (i.e. B^T), ldb.
// Block = 256 thr = 8 waves, block tile (64*MT) x 64, wave tile (16*MT) x 32.
// WMMA fragment layouts used here (bf16 16x16x32, same contiguous k order for A and B):
//   A: lane l, elem e -> m=l&15,  k = e + 16*(l>>4)
//   B: lane l, elem e -> n=l&15,  k = e + 16*(l>>4)
//   C: lane l, vgpr r -> n=l&15,  m = r + 8*(l>>4)
// ---------------------------------------------------------------------------
template<int MT, bool GELU_EP, bool OUT_BF16>
__global__ __launch_bounds__(256)
void gemm_bf16_wmma(const bf16* __restrict__ A, int lda, long saOut, long saIn,
                    const bf16* __restrict__ Bt, int ldb, long sbOut, long sbIn,
                    const float* __restrict__ bias,
                    void* __restrict__ Cp, int ldc, long scOut, long scIn,
                    int M, int Nn, int K, int innerCnt)
{
    int z  = blockIdx.z;
    int zo = z / innerCnt, zi = z % innerCnt;
    A  += zo * saOut + zi * saIn;
    Bt += zo * sbOut + zi * sbIn;
    long coff = zo * scOut + zi * scIn;
    float* Cf = (float*)Cp + coff;
    bf16*  Cb = (bf16*)Cp + coff;

    int m0   = blockIdx.y * (64 * MT);
    int n0   = blockIdx.x * 64;
    int w    = threadIdx.x >> 5;
    int lane = threadIdx.x & 31;
    int wr   = w >> 1;              // 0..3
    int wc   = w & 1;               // 0..1
    int wm   = m0 + wr * (16 * MT);
    int wn   = n0 + wc * 32;
    int lm   = lane & 15;
    int lh   = lane >> 4;

    v8f zero = {0.f,0.f,0.f,0.f,0.f,0.f,0.f,0.f};
    v8f acc[MT][2];
#pragma unroll
    for (int i = 0; i < MT; i++) { acc[i][0] = zero; acc[i][1] = zero; }

    const bf16* aRow[MT];
#pragma unroll
    for (int i = 0; i < MT; i++)
        aRow[i] = A + (long)(wm + i*16 + lm) * lda + lh * 16;
    const bf16* bRow0 = Bt + (long)(wn + lm) * ldb + lh * 16;
    const bf16* bRow1 = Bt + (long)(wn + 16 + lm) * ldb + lh * 16;

    for (int k = 0; k < K; k += 32) {
        Frag a[MT], b0, b1;
#pragma unroll
        for (int i = 0; i < MT; i++) {
            a[i].q[0] = *(const u32x4*)(aRow[i] + k);
            a[i].q[1] = *(const u32x4*)(aRow[i] + k + 8);
        }
        b0.q[0] = *(const u32x4*)(bRow0 + k);
        b0.q[1] = *(const u32x4*)(bRow0 + k + 8);
        b1.q[0] = *(const u32x4*)(bRow1 + k);
        b1.q[1] = *(const u32x4*)(bRow1 + k + 8);
#pragma unroll
        for (int i = 0; i < MT; i++) {
            acc[i][0] = __builtin_amdgcn_wmma_f32_16x16x32_bf16(
                false, a[i].v, false, b0.v, (short)0, acc[i][0], false, false);
            acc[i][1] = __builtin_amdgcn_wmma_f32_16x16x32_bf16(
                false, a[i].v, false, b1.v, (short)0, acc[i][1], false, false);
        }
    }

#pragma unroll
    for (int i = 0; i < MT; i++) {
#pragma unroll
        for (int j = 0; j < 2; j++) {
            int col = wn + j*16 + lm;
            float bv = bias ? bias[col] : 0.f;
#pragma unroll
            for (int r = 0; r < 8; r++) {
                int row = wm + i*16 + r + 8*lh;
                float v = acc[i][j][r] + bv;
                if (GELU_EP) v = gelu_f(v);
                if (OUT_BF16) Cb[(long)row * ldc + col] = (bf16)v;
                else          Cf[(long)row * ldc + col] = v;
            }
        }
    }
}

// ---------------------------------------------------------------------------
// Weight transpose+convert: out[c*R + r] = (bf16) in[r*C + c]   (in: [R,C] f32)
// ---------------------------------------------------------------------------
__global__ __launch_bounds__(256)
void transpose_f32_bf16(const float* __restrict__ in, bf16* __restrict__ out,
                        int R, int C)
{
    __shared__ float tile[32][33];
    int c0 = blockIdx.x * 32, r0 = blockIdx.y * 32;
    int tx = threadIdx.x & 31, ty = threadIdx.x >> 5;  // 32 x 8
#pragma unroll
    for (int i = 0; i < 4; i++) {
        int r = r0 + ty + i*8;
        tile[ty + i*8][tx] = in[(long)r * C + c0 + tx];
    }
    __syncthreads();
#pragma unroll
    for (int i = 0; i < 4; i++) {
        int c = c0 + ty + i*8;
        out[(long)c * R + r0 + tx] = (bf16)tile[tx][ty + i*8];
    }
}

// ---------------------------------------------------------------------------
// fx_mid [SZ, D] (bhnc packed as b,n,h,c) -> FXT [B,H,Dh,N] bf16
// ---------------------------------------------------------------------------
__global__ __launch_bounds__(256)
void transpose_bhnc(const bf16* __restrict__ in, bf16* __restrict__ out)
{
    __shared__ bf16 tile[64][65];
    int z  = blockIdx.y;            // b*H + h
    int b  = z >> 3, h = z & 7;
    int n0 = blockIdx.x * 64;
    int tx = threadIdx.x & 63;
    int ty = threadIdx.x >> 6;      // 0..3
#pragma unroll 4
    for (int i = 0; i < 16; i++) {
        int n = i*4 + ty;
        tile[n][tx] = in[((long)(b*NN + n0 + n)) * DD + h*DHH + tx];
    }
    __syncthreads();
#pragma unroll 4
    for (int i = 0; i < 16; i++) {
        int c = i*4 + ty;
        out[((long)(z*DHH + c)) * NN + n0 + tx] = tile[tx][c];
    }
}

// ---------------------------------------------------------------------------
// spatial bias: sb[b,n,g] = gelu(xy @ sbW1 + sbb1) @ sbW2 + sbb2
// ---------------------------------------------------------------------------
__global__ __launch_bounds__(256)
void spatial_bias(const float* __restrict__ xy,
                  const float* __restrict__ W1, const float* __restrict__ b1,
                  const float* __restrict__ W2, const float* __restrict__ b2,
                  float* __restrict__ sb)
{
    long idx = (long)blockIdx.x * 256 + threadIdx.x;   // (b,n)
    float x0 = xy[idx*2], x1 = xy[idx*2 + 1];
    float hid[32];
#pragma unroll
    for (int i = 0; i < 32; i++)
        hid[i] = gelu_f(x0 * W1[i] + x1 * W1[32 + i] + b1[i]);
    for (int g = 0; g < 64; g++) {
        float s = b2[g];
#pragma unroll
        for (int i = 0; i < 32; i++) s += hid[i] * W2[i*64 + g];
        sb[idx*64 + g] = s;
    }
}

// ---------------------------------------------------------------------------
// Row LayerNorm over D=512 (optionally residual, optional chained 2nd LN).
// ---------------------------------------------------------------------------
__device__ __forceinline__ float block_sum(float v, float* sdata)
{
    int t = threadIdx.x;
    sdata[t] = v; __syncthreads();
    for (int s = 128; s > 0; s >>= 1) {
        if (t < s) sdata[t] += sdata[t + s];
        __syncthreads();
    }
    float r = sdata[0]; __syncthreads();
    return r;
}

__global__ __launch_bounds__(256)
void ln_kernel(const float* __restrict__ x, const float* __restrict__ res,
               const float* __restrict__ g1, const float* __restrict__ be1,
               float* __restrict__ outF, bf16* __restrict__ outB,
               const float* __restrict__ g2, const float* __restrict__ be2,
               bf16* __restrict__ out2)
{
    __shared__ float sdata[256];
    long base = (long)blockIdx.x * DD;
    int  t = threadIdx.x;
    float x0 = x[base + t], x1 = x[base + t + 256];
    if (res) { x0 += res[base + t]; x1 += res[base + t + 256]; }
    float mean = block_sum(x0 + x1, sdata) * (1.f/DD);
    float var  = block_sum(x0*x0 + x1*x1, sdata) * (1.f/DD) - mean*mean;
    float r = rsqrtf(var + 1e-5f);
    float h0 = (x0 - mean) * r * g1[t]     + be1[t];
    float h1 = (x1 - mean) * r * g1[t+256] + be1[t+256];
    if (outF) { outF[base + t] = h0; outF[base + t + 256] = h1; }
    if (outB) { outB[base + t] = (bf16)h0; outB[base + t + 256] = (bf16)h1; }
    if (g2) {
        float m2 = block_sum(h0 + h1, sdata) * (1.f/DD);
        float v2 = block_sum(h0*h0 + h1*h1, sdata) * (1.f/DD) - m2*m2;
        float r2 = rsqrtf(v2 + 1e-5f);
        out2[base + t]       = (bf16)((h0 - m2) * r2 * g2[t]     + be2[t]);
        out2[base + t + 256] = (bf16)((h1 - m2) * r2 * g2[t+256] + be2[t+256]);
    }
}

// ---------------------------------------------------------------------------
// slice softmax: one wave per (b,h,n); writes sw [B,H,N,G] and swT [B,H,G,N]
// ---------------------------------------------------------------------------
__global__ __launch_bounds__(256)
void slice_softmax(const float* __restrict__ logits, const float* __restrict__ sb,
                   const float* __restrict__ temp,
                   bf16* __restrict__ sw, bf16* __restrict__ swT)
{
    int wid = threadIdx.x >> 5, lane = threadIdx.x & 31;
    long idx = (long)blockIdx.x * 8 + wid;            // ((b*H)+h)*N + n
    int n = (int)(idx & (NN - 1));
    int h = (int)((idx >> 13) & 7);
    int b = (int)(idx >> 16);
    float tinv = 1.f / temp[h];
    long lbase = idx * GG;
    long sbase = ((long)b * NN + n) * GG;
    float l0 = logits[lbase + lane]      * tinv + 0.1f * sb[sbase + lane];
    float l1 = logits[lbase + lane + 32] * tinv + 0.1f * sb[sbase + lane + 32];
    float m = fmaxf(l0, l1);
    for (int o = 16; o > 0; o >>= 1) m = fmaxf(m, __shfl_xor(m, o, 32));
    float e0 = expf(l0 - m), e1 = expf(l1 - m);
    float s = e0 + e1;
    for (int o = 16; o > 0; o >>= 1) s += __shfl_xor(s, o, 32);
    float inv = 1.f / s;
    float w0 = e0 * inv, w1 = e1 * inv;
    sw[lbase + lane]      = (bf16)w0;
    sw[lbase + lane + 32] = (bf16)w1;
    long tb = ((long)(b*HH + h) * GG) * NN + n;
    swT[tb + (long)lane * NN]        = (bf16)w0;
    swT[tb + (long)(lane + 32) * NN] = (bf16)w1;
}

// snorm[b,h,g] = sum_n swT row  (one wave per row, fixed-order -> deterministic)
__global__ __launch_bounds__(256)
void snorm_kernel(const bf16* __restrict__ swT, float* __restrict__ snorm)
{
    int wid = threadIdx.x >> 5, lane = threadIdx.x & 31;
    int row = blockIdx.x * 8 + wid;                   // 0 .. B*H*G-1
    const bf16* p = swT + (long)row * NN;
    float s = 0.f;
    for (int i = lane; i < NN; i += 32) s += (float)p[i];
    for (int o = 16; o > 0; o >>= 1) s += __shfl_xor(s, o, 32);
    if (lane == 0) snorm[row] = s;
}

// ---------------------------------------------------------------------------
// Slice-token attention (per batch b; 64x64 matrices, pure VALU, ~50 MFLOP)
// stT: [B,H,Dh,G] f32 (un-normalized), outstT: [B,H,Dh,G] bf16 (= out_st^T)
// ---------------------------------------------------------------------------
__global__ __launch_bounds__(256)
void slice_attn(const float* __restrict__ stT, const float* __restrict__ snorm,
                const float* __restrict__ Wq, const float* __restrict__ Wk,
                const float* __restrict__ Wv,
                const float* __restrict__ srs_p, const float* __restrict__ attn_scale,
                bf16* __restrict__ outstT)
{
    __shared__ float smem[16384];                 // 64 KB
    float* kk = smem;                             // [64][64] normalized k
    float* vv = smem + 4096;                      // [64][64]
    float* qq = smem + 8192;                      // [64][64]; phase1: kv
    float* aw = smem + 12288;                     // [64][64]; phase1: rk[64]
    float* rk = aw;
    int b = blockIdx.x, t = threadIdx.x;
    const float* stB = stT + (long)b * HH * DHH * GG;
    const float* snB = snorm + b * HH * GG;

    // kv = mean_h st/(snorm+eps)    (into qq)
    for (int e = t; e < 4096; e += 256) {
        int g = e >> 6, c = e & 63;
        float s = 0.f;
        for (int h = 0; h < HH; h++)
            s += stB[(h*DHH + c)*GG + g] / (snB[h*GG + g] + 1e-5f);
        qq[e] = s * 0.125f;
    }
    __syncthreads();
    // k = kv@Wk, v = kv@Wv
    for (int e = t; e < 4096; e += 256) {
        int g = e >> 6, c = e & 63;
        float sk = 0.f, sv = 0.f;
        for (int j = 0; j < 64; j++) {
            float kv = qq[g*64 + j];
            sk += kv * Wk[j*64 + c];
            sv += kv * Wv[j*64 + c];
        }
        kk[e] = sk; vv[e] = sv;
    }
    __syncthreads();
    if (t < 64) {
        float s = 0.f;
        for (int c = 0; c < 64; c++) { float x = kk[t*64 + c]; s += x*x; }
        rk[t] = 1.f / fmaxf(sqrtf(s), 1e-12f);
    }
    __syncthreads();
    for (int e = t; e < 4096; e += 256) kk[e] *= rk[e >> 6];
    __syncthreads();

    float srs = *srs_p;
    for (int h = 0; h < HH; h++) {
        const float* stH = stB + h * DHH * GG;
        const float* snH = snB + h * GG;
        // q = stn @ Wq
        for (int e = t; e < 4096; e += 256) {
            int g = e >> 6, c = e & 63;
            float inv = 1.f / (snH[g] + 1e-5f);
            float s = 0.f;
            for (int j = 0; j < 64; j++) s += stH[j*GG + g] * Wq[j*64 + c];
            qq[e] = s * inv;
        }
        __syncthreads();
        // raw logits = q . kn   (q-row norm folded into softmax pass)
        for (int e = t; e < 4096; e += 256) {
            int g = e >> 6, j = e & 63;
            float s = 0.f;
            for (int c = 0; c < 64; c++) s += qq[g*64 + c] * kk[j*64 + c];
            aw[e] = s;
        }
        __syncthreads();
        float scale = attn_scale[h];
        if (t < 64) {
            int g = t;
            float s = 0.f;
            for (int c = 0; c < 64; c++) { float x = qq[g*64 + c]; s += x*x; }
            float mul = scale / fmaxf(sqrtf(s), 1e-12f);
            float m = -3.4e38f;
            for (int j = 0; j < 64; j++) {
                float l = aw[g*64 + j] * mul; aw[g*64 + j] = l;
                if (l > m) m = l;
            }
            float ss = 0.f;
            for (int j = 0; j < 64; j++) {
                float e2 = expf(aw[g*64 + j] - m); aw[g*64 + j] = e2; ss += e2;
            }
            float inv = 1.f / ss;
            for (int j = 0; j < 64; j++) aw[g*64 + j] *= inv;
        }
        __syncthreads();
        // out_st = aw@v + srs*stn  -> stored transposed [c][g] as bf16
        for (int e = t; e < 4096; e += 256) {
            int g = e >> 6, c = e & 63;
            float s = 0.f;
            for (int j = 0; j < 64; j++) s += aw[g*64 + j] * vv[j*64 + c];
            float stn = stH[c*GG + g] / (snH[g] + 1e-5f);
            outstT[((long)(b*HH + h) * DHH + c) * GG + g] = (bf16)(s + srs * stn);
        }
        __syncthreads();
    }
}

// ---------------------------------------------------------------------------
// SE gate
// ---------------------------------------------------------------------------
__global__ __launch_bounds__(256)
void se_stage1(const float* __restrict__ h2, float* __restrict__ part)
{
    int d = blockIdx.x * 256 + threadIdx.x;
    int p = blockIdx.y, b = blockIdx.z;
    const float* base = h2 + ((long)b * NN + p * 512) * DD + d;
    float s = 0.f;
    for (int r = 0; r < 512; r++) s += base[(long)r * DD];
    part[((long)b * 16 + p) * DD + d] = s;
}

__global__ __launch_bounds__(256)
void se_dense(const float* __restrict__ part,
              const float* __restrict__ W1, const float* __restrict__ b1,
              const float* __restrict__ W2, const float* __restrict__ b2,
              float* __restrict__ sev)
{
    __shared__ float sein[DD];
    __shared__ float hid[128];
    int b = blockIdx.x, t = threadIdx.x;
    for (int d = t; d < DD; d += 256) {
        float s = 0.f;
        for (int p = 0; p < 16; p++) s += part[((long)b*16 + p)*DD + d];
        sein[d] = s * (1.f / NN);
    }
    __syncthreads();
    if (t < 128) {
        float s = b1[t];
        for (int d = 0; d < DD; d++) s += sein[d] * W1[d*128 + t];
        hid[t] = gelu_f(s);
    }
    __syncthreads();
    for (int d = t; d < DD; d += 256) {
        float s = b2[d];
        for (int j = 0; j < 128; j++) s += hid[j] * W2[j*DD + d];
        sev[b*DD + d] = 1.f / (1.f + expf(-s));
    }
}

__global__ __launch_bounds__(256)
void se_apply(float* __restrict__ out, const float* __restrict__ sev)
{
    long idx = (long)blockIdx.x * 256 + threadIdx.x;
    int b = (int)(idx / ((long)NN * DD));
    int d = (int)(idx & (DD - 1));
    out[idx] *= sev[b*DD + d];
}

// ===========================================================================
// Host launcher
// ===========================================================================
extern "C" void kernel_launch(void* const* d_in, const int* in_sizes, int n_in,
                              void* d_out, int out_size, void* d_ws, size_t ws_size,
                              hipStream_t stream)
{
    (void)in_sizes; (void)n_in; (void)out_size; (void)ws_size;
    const float* fx      = (const float*)d_in[0];
    const float* raw_xy  = (const float*)d_in[1];
    const float* ln1_g   = (const float*)d_in[2];
    const float* ln1_b   = (const float*)d_in[3];
    const float* Wfx     = (const float*)d_in[4];
    const float* bfx     = (const float*)d_in[5];
    const float* Wx      = (const float*)d_in[6];
    const float* bx      = (const float*)d_in[7];
    const float* Wslice  = (const float*)d_in[8];
    const float* bslice  = (const float*)d_in[9];
    const float* temp    = (const float*)d_in[10];
    const float* Wq      = (const float*)d_in[11];
    const float* Wk      = (const float*)d_in[12];
    const float* Wv      = (const float*)d_in[13];
    const float* srs     = (const float*)d_in[14];
    const float* ascale  = (const float*)d_in[15];
    const float* Wo      = (const float*)d_in[16];
    const float* bo      = (const float*)d_in[17];
    const float* ln1p_g  = (const float*)d_in[18];
    const float* ln1p_b  = (const float*)d_in[19];
    const float* ln2_g   = (const float*)d_in[20];
    const float* ln2_b   = (const float*)d_in[21];
    const float* W1      = (const float*)d_in[22];
    const float* b1      = (const float*)d_in[23];
    const float* W2      = (const float*)d_in[24];
    const float* b2      = (const float*)d_in[25];
    const float* ln2p_g  = (const float*)d_in[26];
    const float* ln2p_b  = (const float*)d_in[27];
    const float* seW1    = (const float*)d_in[28];
    const float* seb1    = (const float*)d_in[29];
    const float* seW2    = (const float*)d_in[30];
    const float* seb2    = (const float*)d_in[31];
    const float* sbW1    = (const float*)d_in[32];
    const float* sbb1    = (const float*)d_in[33];
    const float* sbW2    = (const float*)d_in[34];
    const float* sbb2    = (const float*)d_in[35];
    float* out = (float*)d_out;
    char*  ws  = (char*)d_ws;

    // Workspace layout (byte offsets). Aliasing follows buffer lifetimes.
    constexpr size_t MBy   = 1024ull * 1024ull;
    constexpr size_t O_XLN   = 0;            // bf16 [SZ,D]   (dies after GEMM1/2)
    constexpr size_t O_FXMID = 32*MBy;       // bf16 [SZ,D]   (dies after FXT transpose)
    constexpr size_t O_ATTN  = 0;            // f32  [SZ,D]   (aliases XLN+FXMID)
    constexpr size_t O_MBUF  = 0;            // f32  [SZ,D]   (aliases ATTN)
    constexpr size_t O_XMID  = 64*MBy;       // bf16 [SZ,D]   (dies after logits GEMM)
    constexpr size_t O_OUTX  = 64*MBy;       // bf16 [SZ,D]   (aliases XMID)
    constexpr size_t O_SW    = 96*MBy;       // bf16 [B,H,N,G]
    constexpr size_t O_SWT   = 128*MBy;      // bf16 [B,H,G,N]
    constexpr size_t O_FXT   = 160*MBy;      // bf16 [B,H,Dh,N]
    constexpr size_t O_LOG   = 192*MBy;      // f32  [B,H,N,G] (dies after softmax)
    constexpr size_t O_HBUF  = 192*MBy;      // f32  [SZ,D]   (aliases LOG)
    constexpr size_t O_MID   = 256*MBy;      // bf16 [SZ,DFF]
    constexpr size_t O_HLN   = 384*MBy;      // bf16 [SZ,D]
    constexpr size_t O_SB    = 416*MBy;      // f32  [B,N,G]
    constexpr size_t O_WFXT  = 424*MBy;
    constexpr size_t O_WXT   = O_WFXT + 512*1024;
    constexpr size_t O_WOT   = O_WXT  + 512*1024;
    constexpr size_t O_W1T   = O_WOT  + 512*1024;     // 2 MB
    constexpr size_t O_W2T   = O_W1T  + 2*MBy;        // 2 MB
    constexpr size_t O_WSLT  = O_W2T  + 2*MBy;        // 8 KB
    constexpr size_t O_STT   = O_WSLT + 64*1024;      // f32 512 KB
    constexpr size_t O_SNORM = O_STT  + 512*1024;     // 8 KB
    constexpr size_t O_OUTST = O_SNORM+ 64*1024;      // bf16 256 KB
    constexpr size_t O_SEP   = O_OUTST+ 256*1024;     // f32 128 KB
    constexpr size_t O_SEV   = O_SEP  + 128*1024;     // f32 8 KB

    bf16*  XLN   = (bf16*)(ws + O_XLN);
    bf16*  FXMID = (bf16*)(ws + O_FXMID);
    float* ATTN  = (float*)(ws + O_ATTN);
    float* MBUF  = (float*)(ws + O_MBUF);
    bf16*  XMID  = (bf16*)(ws + O_XMID);
    bf16*  OUTX  = (bf16*)(ws + O_OUTX);
    bf16*  SWb   = (bf16*)(ws + O_SW);
    bf16*  SWT   = (bf16*)(ws + O_SWT);
    bf16*  FXT   = (bf16*)(ws + O_FXT);
    float* LOGI  = (float*)(ws + O_LOG);
    float* HBUF  = (float*)(ws + O_HBUF);
    bf16*  MID   = (bf16*)(ws + O_MID);
    bf16*  HLN   = (bf16*)(ws + O_HLN);
    float* SB    = (float*)(ws + O_SB);
    bf16*  WFXT  = (bf16*)(ws + O_WFXT);
    bf16*  WXT   = (bf16*)(ws + O_WXT);
    bf16*  WOT   = (bf16*)(ws + O_WOT);
    bf16*  W1T   = (bf16*)(ws + O_W1T);
    bf16*  W2T   = (bf16*)(ws + O_W2T);
    bf16*  WSLT  = (bf16*)(ws + O_WSLT);
    float* STT   = (float*)(ws + O_STT);
    float* SNORM = (float*)(ws + O_SNORM);
    bf16*  OUTST = (bf16*)(ws + O_OUTST);
    float* SEP   = (float*)(ws + O_SEP);
    float* SEV   = (float*)(ws + O_SEV);

    dim3 blk(256);

    // 0) weight transpose + f32->bf16 (weights become L2-resident Bt operands)
    transpose_f32_bf16<<<dim3(DD/32,  DD/32),  blk, 0, stream>>>(Wfx,    WFXT, DD,  DD);
    transpose_f32_bf16<<<dim3(DD/32,  DD/32),  blk, 0, stream>>>(Wx,     WXT,  DD,  DD);
    transpose_f32_bf16<<<dim3(DD/32,  DD/32),  blk, 0, stream>>>(Wo,     WOT,  DD,  DD);
    transpose_f32_bf16<<<dim3(DFF/32, DD/32),  blk, 0, stream>>>(W1,     W1T,  DD,  DFF);
    transpose_f32_bf16<<<dim3(DD/32,  DFF/32), blk, 0, stream>>>(W2,     W2T,  DFF, DD);
    transpose_f32_bf16<<<dim3(GG/32,  DHH/32), blk, 0, stream>>>(Wslice, WSLT, DHH, GG);

    // 1) spatial bias
    spatial_bias<<<SZ/256, blk, 0, stream>>>(raw_xy, sbW1, sbb1, sbW2, sbb2, SB);

    // 2) x = LN(fx)  -> bf16
    ln_kernel<<<SZ, blk, 0, stream>>>(fx, nullptr, ln1_g, ln1_b,
                                      nullptr, XLN, nullptr, nullptr, nullptr);

    // 3) fx_mid = x@Wfx + bfx ; x_mid = x@Wx + bx   (bf16 out)
    gemm_bf16_wmma<2,false,true><<<dim3(DD/64, SZ/128, 1), blk, 0, stream>>>(
        XLN, DD, 0, 0, WFXT, DD, 0, 0, bfx, FXMID, DD, 0, 0, SZ, DD, DD, 1);
    gemm_bf16_wmma<2,false,true><<<dim3(DD/64, SZ/128, 1), blk, 0, stream>>>(
        XLN, DD, 0, 0, WXT,  DD, 0, 0, bx,  XMID,  DD, 0, 0, SZ, DD, DD, 1);

    // 4) slice logits (batched over b,h): x_mid[b,:,h,:] @ Wslice + bslice -> f32
    gemm_bf16_wmma<2,false,false><<<dim3(1, NN/128, BB*HH), blk, 0, stream>>>(
        XMID, DD, (long)NN*DD, DHH,
        WSLT, DHH, 0, 0, bslice,
        LOGI, GG, (long)HH*NN*GG, (long)NN*GG,
        NN, GG, DHH, HH);

    // 5) softmax (+temp, +0.1*sb) -> sw (bf16) and swT (bf16)
    slice_softmax<<<(BB*HH*NN)/8, blk, 0, stream>>>(LOGI, SB, temp, SWb, SWT);

    // 6) snorm = sum_n sw
    snorm_kernel<<<(BB*HH*GG)/8, blk, 0, stream>>>(SWT, SNORM);

    // 7) fx_mid -> FXT [B,H,Dh,N]
    transpose_bhnc<<<dim3(NN/64, BB*HH), blk, 0, stream>>>(FXMID, FXT);

    // 8) st^T = fx_mid^T @ sw  (per b,h): [Dh,N]x[N,G] -> f32 [Dh,G]
    gemm_bf16_wmma<1,false,false><<<dim3(1, 1, BB*HH), blk, 0, stream>>>(
        FXT, NN, (long)DHH*NN, 0,
        SWT, NN, (long)GG*NN, 0, nullptr,
        STT, GG, (long)DHH*GG, 0,
        DHH, GG, NN, 1);

    // 9) slice-token attention (q/k/v + softmax + srs residual) -> out_st^T bf16
    slice_attn<<<BB, blk, 0, stream>>>(STT, SNORM, Wq, Wk, Wv, srs, ascale, OUTST);

    // 10) out_x = sw @ out_st (per b,h), scattered into [B,N,D] bf16
    gemm_bf16_wmma<2,false,true><<<dim3(1, NN/128, BB*HH), blk, 0, stream>>>(
        SWb, GG, (long)HH*NN*GG, (long)NN*GG,
        OUTST, GG, (long)HH*DHH*GG, (long)DHH*GG, nullptr,
        OUTX, DD, (long)NN*DD, DHH,
        NN, DHH, GG, HH);

    // 11) attn_out = out_x @ Wo + bo  -> f32
    gemm_bf16_wmma<2,false,false><<<dim3(DD/64, SZ/128, 1), blk, 0, stream>>>(
        OUTX, DD, 0, 0, WOT, DD, 0, 0, bo, ATTN, DD, 0, 0, SZ, DD, DD, 1);

    // 12) h = LN(attn_out + fx, ln1p) -> f32 ; hln2 = LN(h, ln2) -> bf16
    ln_kernel<<<SZ, blk, 0, stream>>>(ATTN, fx, ln1p_g, ln1p_b,
                                      HBUF, nullptr, ln2_g, ln2_b, HLN);

    // 13) MLP: gelu(hln2@W1 + b1) -> bf16 ; @W2 + b2 -> f32
    gemm_bf16_wmma<2,true,true><<<dim3(DFF/64, SZ/128, 1), blk, 0, stream>>>(
        HLN, DD, 0, 0, W1T, DD, 0, 0, b1, MID, DFF, 0, 0, SZ, DFF, DD, 1);
    gemm_bf16_wmma<2,false,false><<<dim3(DD/64, SZ/128, 1), blk, 0, stream>>>(
        MID, DFF, 0, 0, W2T, DFF, 0, 0, b2, MBUF, DD, 0, 0, SZ, DD, DFF, 1);

    // 14) h2 = LN(m + h, ln2p) -> d_out (f32)
    ln_kernel<<<SZ, blk, 0, stream>>>(MBUF, HBUF, ln2p_g, ln2p_b,
                                      out, nullptr, nullptr, nullptr, nullptr);

    // 15) SE gate: mean over n -> gelu fc -> sigmoid fc -> scale d_out
    se_stage1<<<dim3(DD/256, 16, BB), blk, 0, stream>>>(out, SEP);
    se_dense<<<BB, blk, 0, stream>>>(SEP, seW1, seb1, seW2, seb2, SEV);
    se_apply<<<(long)SZ*DD/256, blk, 0, stream>>>(out, SEV);
}
